// SlidingWindowAttention_37864431681668
// MI455X (gfx1250) — matrix operand order repaired and checked by reading, not compiled, not run on hardware
//
#include <hip/hip_runtime.h>
#include <cstdint>

// ---------------------------------------------------------------------------
// SlidingWindowAttention on MI455X (gfx1250), wave32 + WMMA bf16x3 + TDM.
//
//   B=1, S=8192, D=1024, H=16, HD=64, WIN=1024 (+-512), fp32 in/out.
//
// Math note: the reference's chunked overlap-add equals plain per-query
// sliding-window attention EXCEPT the final 128 rows, which get an extra
// blend weight W(p) = 1 + (p-8064)/127 (chunk cs=7168 has ce==S, so no
// trailing ramp, while chunk cs=8064 still ramps up).
//
// Precision: every fp32 operand is split f = hi(bf16) + lo(bf16); products
// use hh + hl + lh (3x v_wmma_f32_16x16x32_bf16) -> ~fp32 accuracy at ~2.7x
// the per-instruction throughput of v_wmma_f32_16x16x4_f32.
//
// Data movement: in the GEMM, all 8 waves of a block share one 32-row A
// panel; it is staged into LDS by the Tensor Data Mover (tensor_load_to_lds,
// double-buffered, TENSORcnt-tracked) while waves stream B from global with
// global_prefetch_b8 lookahead.
// ---------------------------------------------------------------------------

typedef __attribute__((ext_vector_type(16))) __bf16        v16bf;
typedef __attribute__((ext_vector_type(8)))  float         v8f;
typedef __attribute__((ext_vector_type(8)))  unsigned int  v8u;
typedef __attribute__((ext_vector_type(4)))  unsigned int  u32x4;
typedef __attribute__((ext_vector_type(8)))  int           i32x8;
typedef __attribute__((ext_vector_type(4)))  int           i32x4;

#define WMMA_BF16(A, B, C) \
  __builtin_amdgcn_wmma_f32_16x16x32_bf16(false, (A), false, (B), (short)0, (C), false, false)

__device__ __forceinline__ unsigned short f2bf_rne(float f) {
  unsigned u = __float_as_uint(f);
  unsigned r = 0x7FFFu + ((u >> 16) & 1u);
  return (unsigned short)((u + r) >> 16);
}
__device__ __forceinline__ void split_hl(float f, unsigned short& h, unsigned short& l) {
  unsigned short hh = f2bf_rne(f);
  float fh = __uint_as_float(((unsigned)hh) << 16);
  h = hh;
  l = f2bf_rne(f - fh);
}

// --- Tensor Data Mover: DMA a 2D bf16 tile (tile_w x tile_h) global->LDS ---
// D# per CDNA5 ISA:
//   group0: count=1 | lds_addr | global_addr[56:0] | type=2
//   group1: data_size=2B, tensor_dim0=row_stride, tensor_dim1=tile_h,
//           tile_dim0=tile_w, tile_dim1=tile_h, dim0_stride=row_stride
// This toolchain exposes the 6-arg builtin:
//   (u32x4 g0, i32x8 g1, i32x4 g2, i32x4 g3, i32x8 extra, i32 cpol)
__device__ __forceinline__ void tdm_load_tile_2d_bf16(
    unsigned lds_byte_off, const unsigned short* gsrc,
    int tile_w, int tile_h, int row_stride_elems) {
  const unsigned long long ga = (unsigned long long)(uintptr_t)gsrc;
  u32x4 g0;
  g0.x = 1u;                                                  // count=1
  g0.y = lds_byte_off;                                        // LDS dest
  g0.z = (unsigned)ga;                                        // gaddr lo
  g0.w = (unsigned)((ga >> 32) & 0x01FFFFFFu) | (2u << 30);   // gaddr hi|type=2
  i32x8 g1;
  g1[0] = 1 << 16;                                            // data_size=2B
  g1[1] = (row_stride_elems & 0xFFFF) << 16;                  // dim0[15:0]
  g1[2] = ((row_stride_elems >> 16) & 0xFFFF) |               // dim0[31:16]
          ((tile_h & 0xFFFF) << 16);                          // dim1[15:0]
  g1[3] = ((tile_h >> 16) & 0xFFFF) |                         // dim1[31:16]
          ((tile_w & 0xFFFF) << 16);                          // tile_dim0
  g1[4] = tile_h & 0xFFFF;                                    // tile_dim1
  g1[5] = row_stride_elems;                                   // dim0_stride lo
  g1[6] = 0;
  g1[7] = 0;
  __builtin_amdgcn_tensor_load_to_lds(g0, g1, (i32x4)0, (i32x4)0, (i32x8)0, 0);
}

// ---------------------------------------------------------------- split -----
__global__ __launch_bounds__(256) void split_kernel(
    const float* __restrict__ src,
    unsigned short* __restrict__ dh, unsigned short* __restrict__ dl, int n) {
  int i = blockIdx.x * blockDim.x + threadIdx.x;
  if (i < n) { split_hl(src[i], dh[i], dl[i]); }
}

// ---------------------------------------------------------------- GEMM ------
// C[M,N] = A[M,K] * B[N,K]^T + bias[N]    (A,B given as bf16 hi/lo pairs)
// MODE 0: fp32 -> Cf[M,N];  MODE 1: hi/lo split -> Ch/Cl[M,N] (Q,K);
// MODE 2: hi/lo split transposed -> Ch/Cl[N,M] (V, for PV B-layout).
// Block = 8 waves sharing one 32-row A panel (TDM-staged, double-buffered);
// each wave owns a 32-col B slab and computes a 32x32 block (2x2 WMMA tiles).
template <int MODE>
__global__ __launch_bounds__(256) void gemm_bf16x3(
    const unsigned short* __restrict__ Ah, const unsigned short* __restrict__ Al,
    const unsigned short* __restrict__ Bh, const unsigned short* __restrict__ Bl,
    const float* __restrict__ bias,
    float* __restrict__ Cf,
    unsigned short* __restrict__ Ch, unsigned short* __restrict__ Cl,
    int M, int N, int K) {
  __shared__ unsigned short sA[2][2][32 * 32];  // [buf][hi/lo][row*32+k]

  const int lane = threadIdx.x & 31;
  const int wave = threadIdx.x >> 5;
  const int nblk = N >> 5;
  const int wid0 = blockIdx.x * (blockDim.x >> 5);
  const int m0   = (wid0 / nblk) << 5;               // block-uniform
  const int n0   = ((wid0 + wave) % nblk) << 5;      // per-wave
  if (m0 >= M) return;
  const int lr = lane & 15;
  const int lh = lane >> 4;

  v8f acc[2][2];
#pragma unroll
  for (int i = 0; i < 2; i++)
#pragma unroll
    for (int j = 0; j < 2; j++)
#pragma unroll
      for (int r = 0; r < 8; r++) acc[i][j][r] = 0.0f;

  // prologue: TDM-stage first A panel (hi+lo)
  if (wave == 0) {
    tdm_load_tile_2d_bf16((unsigned)(uintptr_t)&sA[0][0][0], Ah + (size_t)m0 * K, 32, 32, K);
    tdm_load_tile_2d_bf16((unsigned)(uintptr_t)&sA[0][1][0], Al + (size_t)m0 * K, 32, 32, K);
  }

  const int nIter = K >> 5;
  for (int it = 0; it < nIter; ++it) {
    const int kk  = it << 5;
    const int buf = it & 1;
    if (wave == 0) __builtin_amdgcn_s_wait_tensorcnt(0);
    __syncthreads();  // publish TDM-written panel; retire prior reads of buf^1
    if (wave == 0 && it + 1 < nIter) {
      tdm_load_tile_2d_bf16((unsigned)(uintptr_t)&sA[buf ^ 1][0][0],
                            Ah + (size_t)m0 * K + kk + 32, 32, 32, K);
      tdm_load_tile_2d_bf16((unsigned)(uintptr_t)&sA[buf ^ 1][1][0],
                            Al + (size_t)m0 * K + kk + 32, 32, 32, K);
    }

    // A tiles from LDS (contiguous k order, same as B: K = 16*half + 2*j)
    v16bf a_h[2], a_l[2], b_h[2], b_l[2];
#pragma unroll
    for (int i = 0; i < 2; i++) {
      const unsigned* ph = (const unsigned*)&sA[buf][0][(i * 16 + lr) * 32];
      const unsigned* pl = (const unsigned*)&sA[buf][1][(i * 16 + lr) * 32];
      v8u uh, ul;
#pragma unroll
      for (int j = 0; j < 8; j++) {
        const int di = 8 * lh + j;
        uh[j] = ph[di];
        ul[j] = pl[di];
      }
      a_h[i] = __builtin_bit_cast(v16bf, uh);
      a_l[i] = __builtin_bit_cast(v16bf, ul);
    }
    // B tiles from global (contiguous k order, same as A: K = 16*half + 2*t) + prefetch
#pragma unroll
    for (int j = 0; j < 2; j++) {
      const size_t off = (size_t)(n0 + j * 16 + lr) * K + kk;
      const unsigned* ph = (const unsigned*)(Bh + off);
      const unsigned* pl = (const unsigned*)(Bl + off);
      v8u uh, ul;
#pragma unroll
      for (int t = 0; t < 8; t++) {
        const int di = 8 * lh + t;
        uh[t] = ph[di];
        ul[t] = pl[di];
      }
      b_h[j] = __builtin_bit_cast(v16bf, uh);
      b_l[j] = __builtin_bit_cast(v16bf, ul);
      if (it + 1 < nIter) {
        __builtin_prefetch(Bh + off + 32, 0, 1);
        __builtin_prefetch(Bl + off + 32, 0, 1);
      }
    }
    // bf16x3: hh + hl + lh
#pragma unroll
    for (int i = 0; i < 2; i++)
#pragma unroll
      for (int j = 0; j < 2; j++) {
        acc[i][j] = WMMA_BF16(a_h[i], b_h[j], acc[i][j]);
        acc[i][j] = WMMA_BF16(a_h[i], b_l[j], acc[i][j]);
        acc[i][j] = WMMA_BF16(a_l[i], b_h[j], acc[i][j]);
      }
  }

  // Epilogue. C/D layout: VGPR r holds row (r + 8*half), col = lane%16.
#pragma unroll
  for (int i = 0; i < 2; i++)
#pragma unroll
    for (int j = 0; j < 2; j++) {
      const int n = n0 + j * 16 + lr;
      const float bv = bias[n];
#pragma unroll
      for (int r = 0; r < 8; r++) {
        const int m = m0 + i * 16 + 8 * lh + r;
        const float v = acc[i][j][r] + bv;
        if constexpr (MODE == 0) {
          Cf[(size_t)m * N + n] = v;
        } else if constexpr (MODE == 1) {
          unsigned short hh, ll;
          split_hl(v, hh, ll);
          Ch[(size_t)m * N + n] = hh;
          Cl[(size_t)m * N + n] = ll;
        } else {
          unsigned short hh, ll;
          split_hl(v, hh, ll);
          Ch[(size_t)n * M + m] = hh;  // transposed: [feature][row]
          Cl[(size_t)n * M + m] = ll;
        }
      }
    }
}

// ------------------------------------------------------------ attention -----
// One wave = one (head, 16-query tile). Flash-style online softmax over the
// sliding window, processed in 32-key blocks. All matmuls are bf16x3 WMMA.
__global__ __launch_bounds__(256) void attn_kernel(
    const unsigned short* __restrict__ Qh, const unsigned short* __restrict__ Ql,
    const unsigned short* __restrict__ Kh, const unsigned short* __restrict__ Kl,
    const unsigned short* __restrict__ Vth, const unsigned short* __restrict__ Vtl,
    unsigned short* __restrict__ Oh, unsigned short* __restrict__ Ol) {
  constexpr int S = 8192, D = 1024, HD = 64, HWIN = 512;
  constexpr float SCALE = 0.125f;  // 1/sqrt(64)
  constexpr float NEG = -3.0e38f;

  __shared__ unsigned short lds_ph[8][16 * 32];  // per-wave P_hi tile
  __shared__ unsigned short lds_pl[8][16 * 32];  // per-wave P_lo tile

  const int lane = threadIdx.x & 31;
  const int wave = threadIdx.x >> 5;
  const int wid  = blockIdx.x * (blockDim.x >> 5) + wave;
  const int head = wid & 15;
  const int q0   = (wid >> 4) << 4;
  const int lr   = lane & 15;
  const int lh   = lane >> 4;

  // --- Q tile in A-layout: [d-half s][hi/lo], 8 VGPRs each -----------------
  v16bf qa[2][2];
  {
    const size_t base = (size_t)(q0 + lr) * D + head * HD;
#pragma unroll
    for (int s = 0; s < 2; s++) {
      const unsigned* ph = (const unsigned*)(Qh + base + s * 32);
      const unsigned* pl = (const unsigned*)(Ql + base + s * 32);
      v8u uh, ul;
#pragma unroll
      for (int j = 0; j < 8; j++) {
        const int di = 8 * lh + j;
        uh[j] = ph[di];
        ul[j] = pl[di];
      }
      qa[s][0] = __builtin_bit_cast(v16bf, uh);
      qa[s][1] = __builtin_bit_cast(v16bf, ul);
    }
  }

  v8f oacc[4];
#pragma unroll
  for (int nt = 0; nt < 4; nt++)
#pragma unroll
    for (int r = 0; r < 8; r++) oacc[nt][r] = 0.0f;
  float mrun[8], lrun[8];
#pragma unroll
  for (int r = 0; r < 8; r++) { mrun[r] = NEG; lrun[r] = 0.0f; }

  int kbeg = q0 - HWIN;
  if (kbeg < 0) kbeg = 0;
  kbeg &= ~31;                // align down; extra keys get masked
  int kend = q0 + 16 + HWIN;  // exclusive
  if (kend > S) kend = S;

  unsigned short* pw_h = lds_ph[wave];
  unsigned short* pw_l = lds_pl[wave];

  for (int kb = kbeg; kb < kend; kb += 32) {
    // --- prefetch next key block's K rows and V panels ---------------------
    const int kbn = kb + 32;
    if (kbn < kend) {
      const int keyp = (kbn + lane) < S ? (kbn + lane) : (S - 1);
      __builtin_prefetch(Kh + (size_t)keyp * D + head * HD, 0, 1);
      __builtin_prefetch(Kl + (size_t)keyp * D + head * HD, 0, 1);
      const size_t vp0 = (size_t)(head * HD + lane) * S + kbn;
      const size_t vp1 = (size_t)(head * HD + 32 + lane) * S + kbn;
      __builtin_prefetch(Vth + vp0, 0, 1);
      __builtin_prefetch(Vth + vp1, 0, 1);
      __builtin_prefetch(Vtl + vp0, 0, 1);
      __builtin_prefetch(Vtl + vp1, 0, 1);
    }

    float p[2][8];
    // --- scores S = (Q . K^T) * scale  (two 16-key subtiles) ---------------
#pragma unroll
    for (int j = 0; j < 2; j++) {
      const int key  = kb + j * 16 + lr;
      const int keyc = key < S ? key : (S - 1);  // clamp addr; value masked
      const size_t base = (size_t)keyc * D + head * HD;
      const unsigned* kh32 = (const unsigned*)(Kh + base);
      const unsigned* kl32 = (const unsigned*)(Kl + base);
      v8f z;
#pragma unroll
      for (int r = 0; r < 8; r++) z[r] = 0.0f;
#pragma unroll
      for (int s = 0; s < 2; s++) {
        v8u uh, ul;
#pragma unroll
        for (int t = 0; t < 8; t++) {
          const int di = s * 16 + 8 * lh + t;  // B-layout within 64-wide head
          uh[t] = kh32[di];
          ul[t] = kl32[di];
        }
        v16bf bh = __builtin_bit_cast(v16bf, uh);
        v16bf bl = __builtin_bit_cast(v16bf, ul);
        z = WMMA_BF16(qa[s][0], bh, z);
        z = WMMA_BF16(qa[s][0], bl, z);
        z = WMMA_BF16(qa[s][1], bh, z);
      }
#pragma unroll
      for (int r = 0; r < 8; r++) {
        const int qabs = q0 + 8 * lh + r;
        const bool ok = (key >= qabs - HWIN) && (key < qabs + HWIN) && (key < S);
        p[j][r] = ok ? z[r] * SCALE : NEG;
      }
    }
    // --- online softmax (row stats live per-VGPR, reduce across 16 lanes) --
#pragma unroll
    for (int r = 0; r < 8; r++) {
      float mx = fmaxf(p[0][r], p[1][r]);
#pragma unroll
      for (int d = 1; d < 16; d <<= 1) mx = fmaxf(mx, __shfl_xor(mx, d, 32));
      const float mnew = fmaxf(mrun[r], mx);
      const float sf = (mrun[r] < -1.0e37f) ? 0.0f : __expf(mrun[r] - mnew);
      const float e0 = (p[0][r] < -1.0e37f) ? 0.0f : __expf(p[0][r] - mnew);
      const float e1 = (p[1][r] < -1.0e37f) ? 0.0f : __expf(p[1][r] - mnew);
      p[0][r] = e0;
      p[1][r] = e1;
      float rs = e0 + e1;
#pragma unroll
      for (int d = 1; d < 16; d <<= 1) rs += __shfl_xor(rs, d, 32);
      lrun[r] = lrun[r] * sf + rs;
      mrun[r] = mnew;
#pragma unroll
      for (int nt = 0; nt < 4; nt++) oacc[nt][r] *= sf;
    }
    // --- C-layout P -> LDS (hi/lo) -> reload in A-layout -------------------
#pragma unroll
    for (int j = 0; j < 2; j++)
#pragma unroll
      for (int r = 0; r < 8; r++) {
        unsigned short hh, ll;
        split_hl(p[j][r], hh, ll);
        const int idx = (8 * lh + r) * 32 + j * 16 + lr;
        pw_h[idx] = hh;
        pw_l[idx] = ll;
      }
    v16bf pa_h, pa_l;
    {
      const unsigned short* rh = pw_h + lr * 32;
      const unsigned short* rl = pw_l + lr * 32;
      v8u uh, ul;
#pragma unroll
      for (int j = 0; j < 8; j++) {
        const int kc = 16 * lh + 2 * j;  // even -> dword
        uh[j] = *(const unsigned*)(rh + kc);
        ul[j] = *(const unsigned*)(rl + kc);
      }
      pa_h = __builtin_bit_cast(v16bf, uh);
      pa_l = __builtin_bit_cast(v16bf, ul);
    }
    // --- O += P . V   (V pre-transposed [h][d][s] -> packed B loads) -------
#pragma unroll
    for (int nt = 0; nt < 4; nt++) {
      const size_t vbase = (size_t)(head * HD + nt * 16 + lr) * S;
      const unsigned* vh32 = (const unsigned*)(Vth + vbase);
      const unsigned* vl32 = (const unsigned*)(Vtl + vbase);
      v8u uh, ul;
#pragma unroll
      for (int j = 0; j < 8; j++) {
        int kx = kb + 16 * lh + 2 * j;
        if (kx > S - 2) kx = S - 2;  // clamp addr; P of masked keys is 0
        uh[j] = vh32[kx >> 1];
        ul[j] = vl32[kx >> 1];
      }
      v16bf bh = __builtin_bit_cast(v16bf, uh);
      v16bf bl = __builtin_bit_cast(v16bf, ul);
      oacc[nt] = WMMA_BF16(pa_h, bh, oacc[nt]);
      oacc[nt] = WMMA_BF16(pa_h, bl, oacc[nt]);
      oacc[nt] = WMMA_BF16(pa_l, bh, oacc[nt]);
    }
  }

  // --- epilogue: normalize, apply reference's tail blend weight, split -----
#pragma unroll
  for (int r = 0; r < 8; r++) {
    const int m = q0 + 8 * lh + r;
    const float wblend =
        (m >= S - 128) ? (1.0f + (float)(m - (S - 128)) * (1.0f / 127.0f)) : 1.0f;
    const float inv = wblend / lrun[r];
#pragma unroll
    for (int nt = 0; nt < 4; nt++) {
      const int n = head * HD + nt * 16 + lr;
      unsigned short hh, ll;
      split_hl(oacc[nt][r] * inv, hh, ll);
      Oh[(size_t)m * D + n] = hh;
      Ol[(size_t)m * D + n] = ll;
    }
  }
}

// ---------------------------------------------------------------- driver ----
extern "C" void kernel_launch(void* const* d_in, const int* in_sizes, int n_in,
                              void* d_out, int out_size, void* d_ws, size_t ws_size,
                              hipStream_t stream) {
  (void)in_sizes; (void)n_in; (void)out_size; (void)ws_size;
  constexpr int S = 8192, D = 1024;
  const size_t SD = (size_t)S * D;
  const size_t DD = (size_t)D * D;

  // setup_inputs() dict order: x, Wq, Wk, Wv, Wo, bq, bk, bv, bo
  const float* x  = (const float*)d_in[0];
  const float* Wq = (const float*)d_in[1];
  const float* Wk = (const float*)d_in[2];
  const float* Wv = (const float*)d_in[3];
  const float* Wo = (const float*)d_in[4];
  const float* bq = (const float*)d_in[5];
  const float* bk = (const float*)d_in[6];
  const float* bv = (const float*)d_in[7];
  const float* bo = (const float*)d_in[8];
  float* out = (float*)d_out;

  // workspace carve-up (all u16 bf16 arrays): ~185 MB total
  unsigned short* w = (unsigned short*)d_ws;
  auto take = [&](size_t n) { unsigned short* r = w; w += n; return r; };
  unsigned short *xh = take(SD), *xl = take(SD);
  unsigned short *wqh = take(DD), *wql = take(DD);
  unsigned short *wkh = take(DD), *wkl = take(DD);
  unsigned short *wvh = take(DD), *wvl = take(DD);
  unsigned short *woh = take(DD), *wol = take(DD);
  unsigned short *qh = take(SD), *ql = take(SD);
  unsigned short *kh = take(SD), *kl = take(SD);
  unsigned short *vth = take(SD), *vtl = take(SD);
  unsigned short *oh = take(SD), *ol = take(SD);

  split_kernel<<<(int)((SD + 255) / 256), 256, 0, stream>>>(x, xh, xl, (int)SD);
  split_kernel<<<(int)((DD + 255) / 256), 256, 0, stream>>>(Wq, wqh, wql, (int)DD);
  split_kernel<<<(int)((DD + 255) / 256), 256, 0, stream>>>(Wk, wkh, wkl, (int)DD);
  split_kernel<<<(int)((DD + 255) / 256), 256, 0, stream>>>(Wv, wvh, wvl, (int)DD);
  split_kernel<<<(int)((DD + 255) / 256), 256, 0, stream>>>(Wo, woh, wol, (int)DD);

  const int gemmBlocks = (S / 32) * (D / 32) / 8;  // 8 waves/block, 32x32/wave
  gemm_bf16x3<1><<<gemmBlocks, 256, 0, stream>>>(xh, xl, wqh, wql, bq, nullptr, qh, ql, S, D, D);
  gemm_bf16x3<1><<<gemmBlocks, 256, 0, stream>>>(xh, xl, wkh, wkl, bk, nullptr, kh, kl, S, D, D);
  gemm_bf16x3<2><<<gemmBlocks, 256, 0, stream>>>(xh, xl, wvh, wvl, bv, nullptr, vth, vtl, S, D, D);

  const int attnBlocks = (16 * (S / 16)) / 8;  // 8192 waves / 8 per block
  attn_kernel<<<attnBlocks, 256, 0, stream>>>(qh, ql, kh, kl, vth, vtl, oh, ol);

  gemm_bf16x3<0><<<gemmBlocks, 256, 0, stream>>>(oh, ol, woh, wol, bo, out, nullptr, nullptr, S, D, D);
}
